// EncoderLayer_70849780515226
// MI455X (gfx1250) — matrix operand order repaired and checked by reading, not compiled, not run on hardware
//
#include <hip/hip_runtime.h>
#include <hip/hip_bf16.h>

// Encoder layer: B=2, S=2048, D=1024, H=16, Dh=64, FF=4096, fp32 ref.
// bf16 WMMA for all GEMMs; flash (streaming-softmax) attention; CDNA5
// async global->LDS staging for bf16 tiles (ASYNCcnt + s_wait_asynccnt);
// ds_swizzle xor-reductions for softmax/LN statistics.

#define DM   1024
#define NH   16
#define DH   64
#define DFF  4096
#define BB   2
#define SS   2048
#define MTOT (BB * SS) // 4096 rows

typedef __attribute__((ext_vector_type(16))) __bf16       v16bf;
typedef __attribute__((ext_vector_type(8)))  float        v8f;
typedef __attribute__((ext_vector_type(4)))  unsigned int u32x4;
typedef __attribute__((ext_vector_type(4)))  float        f32x4;

union Frag16 { v16bf v; u32x4 q[2]; unsigned short u[16]; };

__device__ inline unsigned short f2bf(float f) {
  union { float f; unsigned u; } c; c.f = f;
  unsigned r = c.u + 0x7FFFu + ((c.u >> 16) & 1u); // RNE to bf16
  return (unsigned short)(r >> 16);
}

__device__ inline v8f zero8() {
  v8f z = {0.f, 0.f, 0.f, 0.f, 0.f, 0.f, 0.f, 0.f};
  return z;
}

__device__ inline v8f wmma_bf16(v16bf a, v16bf b, v8f c) {
  return __builtin_amdgcn_wmma_f32_16x16x32_bf16(false, a, false, b,
                                                 (short)0, c, false, false);
}

// ---- CDNA5 async global->LDS (GVS mode), tracked by ASYNCcnt ----
__device__ inline void async_lds_b128(const void* lds_dst, unsigned gbyte_off,
                                      const void* gbase) {
  unsigned lds_addr = (unsigned)(unsigned long long)lds_dst; // low 32b = LDS offset
  unsigned long long base = (unsigned long long)gbase;
  asm volatile("global_load_async_to_lds_b128 %0, %1, %2"
               :: "v"(lds_addr), "v"(gbyte_off), "s"(base)
               : "memory");
}
__device__ inline void wait_async0() {
  asm volatile("s_wait_asynccnt 0x0" ::: "memory");
}

// ---- xor-lane reductions via ds_swizzle (group-of-32: offset=(xor<<10)|0x1F)
template <int XMASK>
__device__ inline float swz_xor(float x) {
  int i = __builtin_bit_cast(int, x);
  i = __builtin_amdgcn_ds_swizzle(i, (XMASK << 10) | 0x1F);
  return __builtin_bit_cast(float, i);
}
// reduce across 16-lane half (columns of a WMMA C fragment)
__device__ inline float half16_max(float x) {
  x = fmaxf(x, swz_xor<1>(x)); x = fmaxf(x, swz_xor<2>(x));
  x = fmaxf(x, swz_xor<4>(x)); x = fmaxf(x, swz_xor<8>(x));
  return x;
}
__device__ inline float half16_sum(float x) {
  x += swz_xor<1>(x); x += swz_xor<2>(x);
  x += swz_xor<4>(x); x += swz_xor<8>(x);
  return x;
}

// A fragment (16x32 bf16) from row-major LDS tile, pitch P ushorts.
// lanes 0-15 -> M=lane, k {0..7,16..23}; lanes 16-31 -> k {8..15,24..31}.
__device__ inline v16bf ldsA(const unsigned short* lds, int row0, int k0, int P, int lane) {
  int hh = lane >> 4, ln = lane & 15;
  Frag16 f;
  const u32x4* p = (const u32x4*)(lds + (row0 + ln) * P + k0 + hh * 8);
  f.q[0] = p[0];
  f.q[1] = p[2];
  return f.v;
}

// B fragment (32x16 bf16) from LDS stored transposed T[n][k], pitch P.
// lanes 0-15 -> N=lane, k {0..7,16..23}; lanes 16-31 -> k {8..15,24..31}.
__device__ inline v16bf ldsB(const unsigned short* lds, int k0, int n0, int P, int lane) {
  int hh = lane >> 4, ln = lane & 15;
  Frag16 f;
  const u32x4* p = (const u32x4*)(lds + (n0 + ln) * P + k0 + hh * 8);
  f.q[0] = p[0];
  f.q[1] = p[2];
  return f.v;
}

// ---------------------------------------------------------------- LayerNorm
__global__ __launch_bounds__(256) void layernorm_bf16(
    const float* __restrict__ x, const float* __restrict__ g,
    const float* __restrict__ beta, unsigned short* __restrict__ out) {
  __shared__ float red[16];
  const int row = blockIdx.x, tid = threadIdx.x;
  const float* xp = x + (size_t)row * DM;
  f32x4 v = *(const f32x4*)(xp + tid * 4);
  float s  = v[0] + v[1] + v[2] + v[3];
  float ss = v[0]*v[0] + v[1]*v[1] + v[2]*v[2] + v[3]*v[3];
  s += swz_xor<1>(s);  ss += swz_xor<1>(ss);
  s += swz_xor<2>(s);  ss += swz_xor<2>(ss);
  s += swz_xor<4>(s);  ss += swz_xor<4>(ss);
  s += swz_xor<8>(s);  ss += swz_xor<8>(ss);
  s += swz_xor<16>(s); ss += swz_xor<16>(ss);
  if ((tid & 31) == 0) { red[tid >> 5] = s; red[8 + (tid >> 5)] = ss; }
  __syncthreads();
  float ts = 0.f, tss = 0.f;
#pragma unroll
  for (int i = 0; i < 8; ++i) { ts += red[i]; tss += red[8 + i]; }
  const float mean = ts * (1.0f / DM);
  const float var  = tss * (1.0f / DM) - mean * mean;
  const float rstd = rsqrtf(var + 1e-5f);
  f32x4 gv = *(const f32x4*)(g + tid * 4);
  f32x4 bv = *(const f32x4*)(beta + tid * 4);
  unsigned short* op = out + (size_t)row * DM + tid * 4;
#pragma unroll
  for (int i = 0; i < 4; ++i) op[i] = f2bf((v[i] - mean) * rstd * gv[i] + bv[i]);
}

// ------------------------------------------------------------------- GEMM
// C[M,N] = A[M,K](bf16) * B[K,N](f32 weights) [+bias][GELU][+res]
// Block tile 128x128, 8 waves (4x2), wave tile 32x64 -> 8 WMMA per K-tile.
// A tile staged via CDNA5 async global->LDS (pure bf16 byte-move); B tile
// converted f32->bf16 on the VALU and stored transposed.
template <bool BIAS, bool GELU, bool RES, bool OUTBF>
__global__ __launch_bounds__(256) void gemm_bf16(
    const unsigned short* __restrict__ A, const float* __restrict__ Bw,
    const float* __restrict__ bias, const float* __restrict__ res,
    void* __restrict__ outp, int M, int N, int K) {
  __shared__ unsigned short As[128 * 40];  // 128x32 bf16, pitch 40
  __shared__ unsigned short Bt[128 * 40];  // B^T: 128(n) x 32(k), pitch 40
  const int tid = threadIdx.x;
  const int lane = tid & 31, w = tid >> 5;
  const int M0 = blockIdx.y * 128, N0 = blockIdx.x * 128;
  const int wm = (w >> 1) * 32, wn = (w & 1) * 64;

  v8f acc[2][4];
#pragma unroll
  for (int mi = 0; mi < 2; ++mi)
#pragma unroll
    for (int t = 0; t < 4; ++t) acc[mi][t] = zero8();

  // staging coordinates
  const int ar = tid >> 1, ac = (tid & 1) * 16;        // A: 128 rows x 32 cols
  const int kr0 = tid >> 5, bc = (tid & 31) * 4;       // B: 32 rows x 128 cols

  for (int k0 = 0; k0 < K; k0 += 32) {
    // async-stage A (128x32 bf16): 2 x b128 per thread
    {
      unsigned goff = (unsigned)(((size_t)(M0 + ar) * K + k0 + ac) * 2);
      async_lds_b128(As + ar * 40 + ac,     goff,      A);
      async_lds_b128(As + ar * 40 + ac + 8, goff + 16, A);
    }
    // stage B^T (32x128 f32 -> bf16): 16 f32 per thread
    {
#pragma unroll
      for (int i = 0; i < 4; ++i) {
        int kr = kr0 + i * 8;
        f32x4 vv = *(const f32x4*)(Bw + (size_t)(k0 + kr) * N + N0 + bc);
#pragma unroll
        for (int j = 0; j < 4; ++j) Bt[(bc + j) * 40 + kr] = f2bf(vv[j]);
      }
      if (k0 + 32 < K)
        __builtin_prefetch((const void*)(Bw + (size_t)(k0 + 32 + kr0) * N + N0 + bc), 0, 1);
    }
    wait_async0();
    __syncthreads();
    v16bf af0 = ldsA(As, wm,      0, 40, lane);
    v16bf af1 = ldsA(As, wm + 16, 0, 40, lane);
#pragma unroll
    for (int t = 0; t < 4; ++t) {
      v16bf bf = ldsB(Bt, 0, wn + 16 * t, 40, lane);
      acc[0][t] = wmma_bf16(af0, bf, acc[0][t]);
      acc[1][t] = wmma_bf16(af1, bf, acc[1][t]);
    }
    __syncthreads();
  }

  const int hh = lane >> 4, ln = lane & 15;
#pragma unroll
  for (int t = 0; t < 4; ++t) {
    const int col = N0 + wn + 16 * t + ln;
    const float bv = BIAS ? bias[col] : 0.0f;
#pragma unroll
    for (int mi = 0; mi < 2; ++mi) {
#pragma unroll
      for (int r = 0; r < 8; ++r) {
        const int row = M0 + wm + mi * 16 + r + 8 * hh;
        float v = acc[mi][t][r] + bv;
        if (GELU) v = 0.5f * v * (1.0f + erff(v * 0.7071067811865475f));
        if (RES) v += res[(size_t)row * N + col];
        if (OUTBF) ((unsigned short*)outp)[(size_t)row * N + col] = f2bf(v);
        else       ((float*)outp)[(size_t)row * N + col] = v;
      }
    }
  }
}

// -------------------------------------------------------- Flash attention
// One block: 128 q-rows of one (b,h); 8 waves, each owns a 16-row strip so
// softmax stats are wave-local. KV streamed in 64-row chunks through LDS.
// K chunk staged via async global->LDS; V transposed manually.
__global__ __launch_bounds__(256) void flash_attn(
    const unsigned short* __restrict__ Q, const unsigned short* __restrict__ Kg,
    const unsigned short* __restrict__ Vg, const float* __restrict__ bias,
    unsigned short* __restrict__ ctx) {
  __shared__ unsigned short Ks[64 * 72];      // K chunk: [kv][dh], pitch 72
  __shared__ unsigned short Vt[64 * 72];      // V chunk transposed: [dh][kv]
  __shared__ unsigned short Ps[8 * 16 * 72];  // per-wave P staging 16x64

  const int tid = threadIdx.x, lane = tid & 31, w = tid >> 5;
  const int hh = lane >> 4, ln = lane & 15;
  const int h = blockIdx.y, b = blockIdx.z;
  const int q0 = blockIdx.x * 128 + w * 16;

  // Q fragments for this wave's 16 rows (K-dim 64 -> two 16x32 frags)
  Frag16 aq0, aq1;
  {
    const unsigned short* qp = Q + ((size_t)(b * SS + q0 + ln)) * DM + h * DH;
    const u32x4* p0 = (const u32x4*)(qp + hh * 8);
    aq0.q[0] = p0[0]; aq0.q[1] = p0[2];
    const u32x4* p1 = (const u32x4*)(qp + 32 + hh * 8);
    aq1.q[0] = p1[0]; aq1.q[1] = p1[2];
  }

  v8f o[4];
#pragma unroll
  for (int t = 0; t < 4; ++t) o[t] = zero8();
  float mrun[8], lrun[8];
#pragma unroll
  for (int r = 0; r < 8; ++r) { mrun[r] = -1e30f; lrun[r] = 0.f; }

  unsigned short* Pw = Ps + w * 16 * 72;
  const int sr = tid >> 2, sc = (tid & 3) * 16;  // staging coords: 64 x 64

  for (int j = 0; j < SS; j += 64) {
    __syncthreads();  // previous chunk fully consumed by all waves
    {
      const unsigned goff =
          (unsigned)(((size_t)(b * SS + j + sr) * DM + h * DH + sc) * 2);
      // K chunk: async byte-move, 2 x b128 per thread
      async_lds_b128(Ks + sr * 72 + sc,     goff,      Kg);
      async_lds_b128(Ks + sr * 72 + sc + 8, goff + 16, Kg);
      // V chunk transposed (needs lane-level transpose -> manual)
      const u32x4* vs = (const u32x4*)(Vg + (size_t)(b * SS + j + sr) * DM + h * DH + sc);
      u32x4 v0 = vs[0], v1 = vs[1];
      const unsigned short* u0 = (const unsigned short*)&v0;
      const unsigned short* u1 = (const unsigned short*)&v1;
#pragma unroll
      for (int i = 0; i < 8; ++i) {
        Vt[(sc + i) * 72 + sr]     = u0[i];
        Vt[(sc + 8 + i) * 72 + sr] = u1[i];
      }
    }
    wait_async0();
    __syncthreads();

    // S = Q * K^T : 16 rows x 64 kv cols per wave
    v8f s[4];
#pragma unroll
    for (int t = 0; t < 4; ++t) {
      v16bf b0 = ldsB(Ks, 0, 16 * t, 72, lane);
      v16bf b1 = ldsB(Ks, 32, 16 * t, 72, lane);
      v8f z = zero8();
      z = wmma_bf16(aq0.v, b0, z);
      z = wmma_bf16(aq1.v, b1, z);
      s[t] = z;
    }
    // scale + relative bias
#pragma unroll
    for (int t = 0; t < 4; ++t) {
      const int kc = j + 16 * t + ln;
      const float* bp = bias + (size_t)h * SS * SS + (size_t)q0 * SS + kc;
#pragma unroll
      for (int r = 0; r < 8; ++r)
        s[t][r] = s[t][r] * 0.125f + bp[(size_t)(r + 8 * hh) * SS];
    }
    // online softmax (ds_swizzle xor-reduce across the 16-lane half)
    float mnew[8], alpha[8];
#pragma unroll
    for (int r = 0; r < 8; ++r) {
      float mx = fmaxf(fmaxf(s[0][r], s[1][r]), fmaxf(s[2][r], s[3][r]));
      mx = half16_max(mx);
      mnew[r]  = fmaxf(mrun[r], mx);
      alpha[r] = __expf(mrun[r] - mnew[r]);
      mrun[r]  = mnew[r];
    }
#pragma unroll
    for (int r = 0; r < 8; ++r) {
      float rs = 0.f;
#pragma unroll
      for (int t = 0; t < 4; ++t) {
        float p = __expf(s[t][r] - mnew[r]);
        s[t][r] = p;
        rs += p;
      }
      rs = half16_sum(rs);
      lrun[r] = lrun[r] * alpha[r] + rs;
    }
#pragma unroll
    for (int t = 0; t < 4; ++t)
#pragma unroll
      for (int r = 0; r < 8; ++r) o[t][r] *= alpha[r];

    // C-layout -> A-layout via per-wave LDS round trip (same-wave DS ordered)
#pragma unroll
    for (int t = 0; t < 4; ++t)
#pragma unroll
      for (int r = 0; r < 8; ++r)
        Pw[(r + 8 * hh) * 72 + 16 * t + ln] = f2bf(s[t][r]);

    Frag16 ap0, ap1;
    {
      const u32x4* p0 = (const u32x4*)(Pw + ln * 72 + hh * 8);
      ap0.q[0] = p0[0]; ap0.q[1] = p0[2];
      const u32x4* p1 = (const u32x4*)(Pw + ln * 72 + 32 + hh * 8);
      ap1.q[0] = p1[0]; ap1.q[1] = p1[2];
    }
    // O += P * V
#pragma unroll
    for (int t = 0; t < 4; ++t) {
      v16bf bv0 = ldsB(Vt, 0, 16 * t, 72, lane);
      v16bf bv1 = ldsB(Vt, 32, 16 * t, 72, lane);
      o[t] = wmma_bf16(ap0.v, bv0, o[t]);
      o[t] = wmma_bf16(ap1.v, bv1, o[t]);
    }
  }

  // normalize and store ctx (bf16, row-major [b*S+s][h*64+dh])
#pragma unroll
  for (int r = 0; r < 8; ++r) lrun[r] = 1.0f / lrun[r];
#pragma unroll
  for (int t = 0; t < 4; ++t)
#pragma unroll
    for (int r = 0; r < 8; ++r) {
      const size_t row = (size_t)(b * SS + q0 + r + 8 * hh);
      ctx[row * DM + h * DH + 16 * t + ln] = f2bf(o[t][r] * lrun[r]);
    }
}

// ----------------------------------------------------------------- driver
extern "C" void kernel_launch(void* const* d_in, const int* in_sizes, int n_in,
                              void* d_out, int out_size, void* d_ws, size_t ws_size,
                              hipStream_t stream) {
  (void)in_sizes; (void)n_in; (void)out_size; (void)ws_size;
  const float* x   = (const float*)d_in[0];
  const float* rb  = (const float*)d_in[1];
  const float* wq  = (const float*)d_in[2];
  const float* wk  = (const float*)d_in[3];
  const float* wv  = (const float*)d_in[4];
  const float* wo  = (const float*)d_in[5];
  const float* wob = (const float*)d_in[6];
  const float* w1  = (const float*)d_in[7];
  const float* b1  = (const float*)d_in[8];
  const float* w2  = (const float*)d_in[9];
  const float* b2  = (const float*)d_in[10];
  const float* g1  = (const float*)d_in[11];
  const float* be1 = (const float*)d_in[12];
  const float* g2  = (const float*)d_in[13];
  const float* be2 = (const float*)d_in[14];
  float* out = (float*)d_out;

  // workspace carve (72 MB total, bf16 tensors)
  char* ws = (char*)d_ws;
  unsigned short* xn  = (unsigned short*)(ws);                              // 8 MB
  unsigned short* q   = (unsigned short*)(ws + (size_t)8  * 1024 * 1024);   // 8 MB
  unsigned short* k   = (unsigned short*)(ws + (size_t)16 * 1024 * 1024);   // 8 MB
  unsigned short* v   = (unsigned short*)(ws + (size_t)24 * 1024 * 1024);   // 8 MB
  unsigned short* ctx = (unsigned short*)(ws + (size_t)32 * 1024 * 1024);   // 8 MB
  unsigned short* hbf = (unsigned short*)(ws + (size_t)40 * 1024 * 1024);   // 32 MB

  const dim3 blk(256);
  const dim3 gq(DM / 128, MTOT / 128);
  const dim3 gff(DFF / 128, MTOT / 128);
  const dim3 gfl(SS / 128, NH, BB);

  layernorm_bf16<<<MTOT, blk, 0, stream>>>(x, g1, be1, xn);
  gemm_bf16<false, false, false, true><<<gq, blk, 0, stream>>>(xn, wq, nullptr, nullptr, q, MTOT, DM, DM);
  gemm_bf16<false, false, false, true><<<gq, blk, 0, stream>>>(xn, wk, nullptr, nullptr, k, MTOT, DM, DM);
  gemm_bf16<false, false, false, true><<<gq, blk, 0, stream>>>(xn, wv, nullptr, nullptr, v, MTOT, DM, DM);
  flash_attn<<<gfl, blk, 0, stream>>>(q, k, v, rb, ctx);
  gemm_bf16<true, false, true, false><<<gq, blk, 0, stream>>>(ctx, wo, wob, x, out, MTOT, DM, DM);
  layernorm_bf16<<<MTOT, blk, 0, stream>>>(out, g2, be2, xn);
  gemm_bf16<true, true, false, true><<<gff, blk, 0, stream>>>(xn, w1, b1, nullptr, hbf, MTOT, DFF, DM);
  gemm_bf16<true, false, true, false><<<gq, blk, 0, stream>>>(hbf, w2, b2, out, out, MTOT, DM, DFF);
}
